// MultichannelMultiheadAttention_4475355922821
// MI455X (gfx1250) — matrix operand order repaired and checked by reading, not compiled, not run on hardware
//
#include <hip/hip_runtime.h>
#include <math.h>

// ---------------------------------------------------------------------------
// MultichannelMultiheadAttention for MI455X (gfx1250, wave32, WMMA).
// B=2 C=8 F=512 W=512 H=8 D=64.
// All GEMMs use v_wmma_f32_16x16x32_bf16 with fully-unrolled K loops
// (compile-time trip count -> no register rotation copies, immediate-offset
// b128 loads, scheduler-pipelined load/WMMA overlap).
// ---------------------------------------------------------------------------

#define B_ 2
#define C_ 8
#define F_ 512
#define W_ 512
#define H_ 8
#define D_ 64

typedef __bf16 bf16;
typedef __attribute__((ext_vector_type(16))) __bf16 v16bf;
typedef __attribute__((ext_vector_type(8)))  __bf16 v8bf;
typedef __attribute__((ext_vector_type(8)))  float  v8f;

// ---- WMMA operand loaders (ISA 7.12.2, 16-bit A 16x32 / B 32x16) ----------
// A: lane = hf*16 + m.  Lane's 16 bf16 = K [16hf .. 16hf+15] (same order as B)
//    -> one contiguous 32-byte load from the row's K-block base.
__device__ __forceinline__ v16bf loadA(const bf16* p, int hf) {
  return *reinterpret_cast<const v16bf*>(p + 16 * hf);
}
// B: lane = hf*16 + n.  Lane's 16 bf16 = K rows [16hf .. 16hf+15] at column n
//    -> one contiguous 32-byte load from the column's K-block base.
__device__ __forceinline__ v16bf loadB(const bf16* p, int hf) {
  return *reinterpret_cast<const v16bf*>(p + 16 * hf);
}

__device__ __forceinline__ v8f wmma_bf16(v16bf a, v16bf b, v8f c) {
  return __builtin_amdgcn_wmma_f32_16x16x32_bf16(false, a, false, b,
                                                 (short)0, c, false, false);
}

// ---------------------------------------------------------------------------
// Kernel 1: x (BC,F,W) fp32 -> xT (BC,W,F) bf16   (LDS 32x32 tile transpose)
// ---------------------------------------------------------------------------
__global__ __launch_bounds__(256)
void transpose_x_kernel(const float* __restrict__ x, bf16* __restrict__ xT) {
  __shared__ float tile[32][33];
  const int bc = blockIdx.z;
  const int f0 = blockIdx.y * 32, w0 = blockIdx.x * 32;
  const float* xb = x + (size_t)bc * (F_ * W_);
  for (int i = threadIdx.y; i < 32; i += 8)
    tile[i][threadIdx.x] = xb[(size_t)(f0 + i) * W_ + w0 + threadIdx.x];
  __syncthreads();
  bf16* xtb = xT + (size_t)bc * (W_ * F_);
  for (int i = threadIdx.y; i < 32; i += 8)
    xtb[(size_t)(w0 + i) * F_ + f0 + threadIdx.x] = (bf16)tile[threadIdx.x][i];
}

// ---------------------------------------------------------------------------
// Kernel 2: Wq|Wk|Wv|Wo fp32 -> contiguous bf16 [4][C][F][F]
// ---------------------------------------------------------------------------
__global__ __launch_bounds__(256)
void convert_w_kernel(const float* __restrict__ Wq, const float* __restrict__ Wk,
                      const float* __restrict__ Wv, const float* __restrict__ Wo,
                      bf16* __restrict__ Wbf) {
  const size_t CFF = (size_t)C_ * F_ * F_;
  const size_t i = (size_t)blockIdx.x * 256 + threadIdx.x;
  const int p = (int)(i / CFF);
  const size_t r = i % CFF;
  const float* s = (p == 0) ? Wq : (p == 1) ? Wk : (p == 2) ? Wv : Wo;
  Wbf[i] = (bf16)s[r];
}

// ---------------------------------------------------------------------------
// Kernel 3: rope table: ropeC/ropeS[w][i] = cos/sin(w * inv_freq[i]),
// 512 x 32 fp32 each (128 KB total -> lives in L2, reused by all 32 GEMMs).
// Keeps the Payne-Hanek sincos slow path out of the hot WMMA loop.
// ---------------------------------------------------------------------------
__global__ __launch_bounds__(256)
void rope_table_kernel(const float* __restrict__ inv_freq,
                       float* __restrict__ ropeC, float* __restrict__ ropeS) {
  const int idx = blockIdx.x * 256 + threadIdx.x;  // idx = w*32 + i
  const int w = idx >> 5, i = idx & 31;
  float sn, cs;
  sincosf((float)w * inv_freq[i], &sn, &cs);
  ropeC[idx] = cs;
  ropeS[idx] = sn;
}

// ---------------------------------------------------------------------------
// Kernel 4: QKV projection.  Y = W[c] @ x[b,c]  (512x512x512, bf16 WMMA),
// rotary (table lookup) applied in-register for Q,K.
// Q,K stored (h,w,d)-transposed; V stored un-transposed (f,w) so the P@V
// kernel's B operand is contiguous.  16(M) x 64(N) per wave, unrolled K.
// g = ((bc*3+proj)*32+mt)*8+nt.
// ---------------------------------------------------------------------------
__global__ __launch_bounds__(128)
void proj_qkv_kernel(const bf16* __restrict__ Wbf, const bf16* __restrict__ xT,
                     const float* __restrict__ ropeC,
                     const float* __restrict__ ropeS,
                     bf16* __restrict__ Qt, bf16* __restrict__ Kt,
                     bf16* __restrict__ Vw) {
  const int g    = blockIdx.x * 4 + (threadIdx.x >> 5);
  const int lane = threadIdx.x & 31;
  const int hf = lane >> 4, sub = lane & 15;
  const int nt   = g & 7;
  const int mt   = (g >> 3) & 31;
  const int proj = (g >> 8) % 3;
  const int bc   = g / 768;
  const int c    = bc % C_;
  const int m0 = mt * 16, nbase = nt * 64;

  const bf16* A  = Wbf + ((size_t)proj * C_ + c) * (F_ * F_) + (size_t)(m0 + sub) * F_;
  const bf16* B0 = xT + (size_t)bc * (W_ * F_) + (size_t)(nbase + sub) * F_;

  v8f acc[4] = {v8f{}, v8f{}, v8f{}, v8f{}};
#pragma unroll
  for (int kb = 0; kb < F_; kb += 32) {
    const v16bf a = loadA(A + kb, hf);
    v16bf b[4];
#pragma unroll
    for (int t = 0; t < 4; ++t) b[t] = loadB(B0 + t * (16 * F_) + kb, hf);
#pragma unroll
    for (int t = 0; t < 4; ++t) acc[t] = wmma_bf16(a, b[t], acc[t]);
  }

  // C/D layout: VGPR r holds M = 8*hf + r, N = sub.  Row f = m0 + 8*hf + r.
  const int head = m0 >> 6;
  const int d0   = (m0 & 63) + 8 * hf;  // 8 consecutive d per lane
  if (proj < 2) {  // Q or K: rotary + (h,w,d)-transposed 16B packed store
    bf16* dst = (proj == 0) ? Qt : Kt;
    const int pr0 = d0 >> 1;  // 4 consecutive rope pairs per lane
#pragma unroll
    for (int t = 0; t < 4; ++t) {
      const int w = nbase + t * 16 + sub;
      const float4 c4 = *reinterpret_cast<const float4*>(ropeC + w * 32 + pr0);
      const float4 s4 = *reinterpret_cast<const float4*>(ropeS + w * 32 + pr0);
      const float cc[4] = {c4.x, c4.y, c4.z, c4.w};
      const float ss[4] = {s4.x, s4.y, s4.z, s4.w};
#pragma unroll
      for (int rp = 0; rp < 4; ++rp) {
        const int r = 2 * rp;
        float x0 = acc[t][r], x1 = acc[t][r + 1];
        acc[t][r]     = x0 * cc[rp] - x1 * ss[rp];
        acc[t][r + 1] = x1 * cc[rp] + x0 * ss[rp];
      }
      v8bf vout;
#pragma unroll
      for (int r = 0; r < 8; ++r) vout[r] = (bf16)acc[t][r];
      *reinterpret_cast<v8bf*>(dst + (((size_t)bc * H_ + head) * W_ + w) * D_ + d0) = vout;
    }
  } else {  // V: un-transposed (bc, f, w)
#pragma unroll
    for (int t = 0; t < 4; ++t) {
      const int w = nbase + t * 16 + sub;
#pragma unroll
      for (int r = 0; r < 8; ++r)
        Vw[(size_t)bc * (F_ * W_) + (size_t)(m0 + 8 * hf + r) * W_ + w] =
            (bf16)acc[t][r];
    }
  }
}

// ---------------------------------------------------------------------------
// Kernel 5: qk = Q Kᵀ / sqrt(512), fp32 into d_out's qk region.
// One wave per 16x16 tile, K=64 -> 2 WMMAs.  g = bch*1024 + it*32 + jt.
// ---------------------------------------------------------------------------
__global__ __launch_bounds__(128)
void qk_kernel(const bf16* __restrict__ Qt, const bf16* __restrict__ Kt,
               float* __restrict__ qk) {
  const int g    = blockIdx.x * 4 + (threadIdx.x >> 5);
  const int lane = threadIdx.x & 31;
  const int hf = lane >> 4, sub = lane & 15;
  const int bch = g >> 10;
  const int i0 = ((g >> 5) & 31) * 16, j0 = (g & 31) * 16;
  const bf16* Qrow = Qt + ((size_t)bch * W_ + i0 + sub) * D_;
  const bf16* Krow = Kt + ((size_t)bch * W_ + j0 + sub) * D_;
  v16bf a0 = loadA(Qrow, hf), a1 = loadA(Qrow + 32, hf);
  v16bf b0 = loadB(Krow, hf), b1 = loadB(Krow + 32, hf);
  v8f acc = {};
  acc = wmma_bf16(a0, b0, acc);
  acc = wmma_bf16(a1, b1, acc);
  const float scale = 0.04419417382415922f;  // 1/sqrt(512)
  float* row = qk + (size_t)bch * (W_ * W_);
#pragma unroll
  for (int r = 0; r < 8; ++r)
    row[(size_t)(i0 + 8 * hf + r) * W_ + (j0 + sub)] = acc[r] * scale;
}

// ---------------------------------------------------------------------------
// Kernel 6: row softmax of qk (512/row), one wave32 per row, probs -> bf16.
// ---------------------------------------------------------------------------
__global__ __launch_bounds__(256)
void softmax_kernel(const float* __restrict__ qk, bf16* __restrict__ P) {
  const int wv   = blockIdx.x * 8 + (threadIdx.x >> 5);
  const int lane = threadIdx.x & 31;
  const float* row = qk + (size_t)wv * W_;
  float v[16];
  float m = -3.4e38f;
#pragma unroll
  for (int t = 0; t < 16; ++t) {
    v[t] = row[lane + 32 * t];
    m = fmaxf(m, v[t]);
  }
#pragma unroll
  for (int off = 16; off > 0; off >>= 1) m = fmaxf(m, __shfl_xor(m, off, 32));
  float s = 0.f;
#pragma unroll
  for (int t = 0; t < 16; ++t) {
    v[t] = __expf(v[t] - m);
    s += v[t];
  }
#pragma unroll
  for (int off = 16; off > 0; off >>= 1) s += __shfl_xor(s, off, 32);
  const float inv = 1.0f / s;
  bf16* pr = P + (size_t)wv * W_;
#pragma unroll
  for (int t = 0; t < 16; ++t) pr[lane + 32 * t] = (bf16)(v[t] * inv);
}

// ---------------------------------------------------------------------------
// Kernel 7: a = P @ V  (512x64x512), unrolled K; aT stored (w,f)-transposed.
// B operand = V rows (d fixed per lane) -> contiguous 32B loads.
// g = bch*128 + it*4 + dt.
// ---------------------------------------------------------------------------
__global__ __launch_bounds__(128)
void av_kernel(const bf16* __restrict__ P, const bf16* __restrict__ Vw,
               bf16* __restrict__ aT) {
  const int g    = blockIdx.x * 4 + (threadIdx.x >> 5);
  const int lane = threadIdx.x & 31;
  const int hf = lane >> 4, sub = lane & 15;
  const int dt = g & 3, it = (g >> 2) & 31, bch = g >> 7;
  const int bc = bch >> 3, head = bch & 7;
  const int i0 = it * 16;
  const int dcol = dt * 16 + sub;
  const bf16* Prow = P + ((size_t)bch * W_ + i0 + sub) * W_;
  const bf16* Vrow = Vw + (size_t)bc * (F_ * W_) + (size_t)(head * D_ + dcol) * W_;

  v8f acc = {};
#pragma unroll
  for (int kb = 0; kb < W_; kb += 32) {
    const v16bf a = loadA(Prow + kb, hf);
    const v16bf b = loadB(Vrow + kb, hf);
    acc = wmma_bf16(a, b, acc);
  }

  bf16* arow = aT + (size_t)bc * (W_ * F_);
  const int f = head * D_ + dcol;
#pragma unroll
  for (int r = 0; r < 8; ++r)
    arow[(size_t)(i0 + 8 * hf + r) * F_ + f] = (bf16)acc[r];
}

// ---------------------------------------------------------------------------
// Kernel 8: out = Wo[c] @ a  (512x512x512), fp32 result, unrolled K.
// g = bc*256 + mt*8 + nt.
// ---------------------------------------------------------------------------
__global__ __launch_bounds__(128)
void out_kernel(const bf16* __restrict__ Wo, const bf16* __restrict__ aT,
                float* __restrict__ out) {
  const int g    = blockIdx.x * 4 + (threadIdx.x >> 5);
  const int lane = threadIdx.x & 31;
  const int hf = lane >> 4, sub = lane & 15;
  const int nt = g & 7, mt = (g >> 3) & 31, bc = g >> 8;
  const int c = bc % C_;
  const int m0 = mt * 16, nbase = nt * 64;
  const bf16* A  = Wo + (size_t)c * (F_ * F_) + (size_t)(m0 + sub) * F_;
  const bf16* B0 = aT + (size_t)bc * (W_ * F_) + (size_t)(nbase + sub) * F_;

  v8f acc[4] = {v8f{}, v8f{}, v8f{}, v8f{}};
#pragma unroll
  for (int kb = 0; kb < F_; kb += 32) {
    const v16bf a = loadA(A + kb, hf);
    v16bf b[4];
#pragma unroll
    for (int t = 0; t < 4; ++t) b[t] = loadB(B0 + t * (16 * F_) + kb, hf);
#pragma unroll
    for (int t = 0; t < 4; ++t) acc[t] = wmma_bf16(a, b[t], acc[t]);
  }

  float* ob = out + (size_t)bc * (F_ * W_);
#pragma unroll
  for (int t = 0; t < 4; ++t) {
    const int w = nbase + t * 16 + sub;
#pragma unroll
    for (int r = 0; r < 8; ++r)
      ob[(size_t)(m0 + 8 * hf + r) * W_ + w] = acc[t][r];
  }
}

// ---------------------------------------------------------------------------
// Host launcher.  Workspace layout (bf16 elements):
//   xT 4Mi | Wbf 8Mi | Qt 4Mi | Kt 4Mi | Vw 4Mi | P 32Mi | aT 4Mi | rope 64Ki f32
// total ~120.1 MiB.
// ---------------------------------------------------------------------------
extern "C" void kernel_launch(void* const* d_in, const int* in_sizes, int n_in,
                              void* d_out, int out_size, void* d_ws, size_t ws_size,
                              hipStream_t stream) {
  (void)in_sizes; (void)n_in; (void)out_size; (void)ws_size;
  const float* x        = (const float*)d_in[0];
  const float* Wq       = (const float*)d_in[1];
  const float* Wk       = (const float*)d_in[2];
  const float* Wv       = (const float*)d_in[3];
  const float* Wo       = (const float*)d_in[4];
  const float* inv_freq = (const float*)d_in[5];

  float* out = (float*)d_out;                    // (B,C,F,W)
  float* qk  = out + (size_t)B_ * C_ * F_ * W_;  // (B,C,H,W,W)

  const size_t XT  = (size_t)B_ * C_ * W_ * F_;                // 4,194,304
  const size_t WBF = (size_t)4 * C_ * F_ * F_;                 // 8,388,608
  const size_t PSZ = (size_t)B_ * C_ * H_ * W_ * (size_t)W_;   // 33,554,432

  bf16* ws  = (bf16*)d_ws;
  bf16* xT  = ws;
  bf16* Wbf = xT + XT;
  bf16* Qt  = Wbf + WBF;
  bf16* Kt  = Qt + XT;
  bf16* Vw  = Kt + XT;
  bf16* P   = Vw + XT;
  bf16* aT  = P + PSZ;
  float* ropeC = (float*)(aT + XT);
  float* ropeS = ropeC + (size_t)W_ * 32;

  transpose_x_kernel<<<dim3(W_ / 32, F_ / 32, B_ * C_), dim3(32, 8), 0, stream>>>(x, xT);
  convert_w_kernel<<<(unsigned)(WBF / 256), 256, 0, stream>>>(Wq, Wk, Wv, Wo, Wbf);
  rope_table_kernel<<<(W_ * 32) / 256, 256, 0, stream>>>(inv_freq, ropeC, ropeS);
  // 16 bc * 3 proj * 32 mt * 8 nt = 12288 waves
  proj_qkv_kernel<<<12288 / 4, 128, 0, stream>>>(Wbf, xT, ropeC, ropeS, Qt, Kt, Vw);
  // 128 bch * 32 * 32 = 131072 waves
  qk_kernel<<<131072 / 4, 128, 0, stream>>>(Qt, Kt, qk);
  // 65536 rows, 8 waves/block
  softmax_kernel<<<65536 / 8, 256, 0, stream>>>(qk, P);
  // 128 bch * 32 it * 4 dt = 16384 waves
  av_kernel<<<16384 / 4, 128, 0, stream>>>(P, Vw, aT);
  // 16 bc * 32 mt * 8 nt = 4096 waves
  out_kernel<<<4096 / 4, 128, 0, stream>>>(Wbf + (size_t)3 * C_ * F_ * F_, aT, out);
}
